// MoE_75660143886816
// MI455X (gfx1250) — matrix operand order repaired and checked by reading, not compiled, not run on hardware
//
#include <hip/hip_runtime.h>
#include <stdint.h>
#include <stddef.h>

// ---------------- problem constants (from reference) ----------------
#define BATCH  4096
#define FEAT   512            // I == H == O
#define NEXP   8
#define NC     8              // GRID_SIZE + ORDER spline coefficients
#define KFUSE  (FEAT * (NC + 1))   // 4608 fused inner dimension
#define SPLORD 3
#define GRIDN  5

// ---------------- WMMA types ----------------
typedef __attribute__((ext_vector_type(16))) __bf16       v16bf;
typedef __attribute__((ext_vector_type(8)))  float        v8f;
typedef __attribute__((ext_vector_type(4)))  unsigned int u32x4;

// float -> bf16, round-to-nearest-even
__device__ __forceinline__ unsigned short f2bf(float f) {
  unsigned int u = __float_as_uint(f);
  unsigned int r = u + 0x7FFFu + ((u >> 16) & 1u);
  return (unsigned short)(r >> 16);
}

// CDNA5 async global->LDS (ASYNCcnt-tracked), 16B per lane
__device__ __forceinline__ void async_b128(unsigned lds_off, const void* gptr) {
  asm volatile("global_load_async_to_lds_b128 %0, %1, off"
               :: "v"(lds_off), "v"((unsigned long long)(uintptr_t)gptr)
               : "memory");
}
__device__ __forceinline__ void wait_async0() {
  asm volatile("s_wait_asynccnt 0x0" ::: "memory");
}

// ---------------- gating: softmax(x @ gate_w^T + gate_b) ----------------
__global__ void gate_kernel(const float* __restrict__ x,
                            const float* __restrict__ gw,
                            const float* __restrict__ gb,
                            float* __restrict__ gate) {
  int b = blockIdx.x * blockDim.x + threadIdx.x;
  if (b >= BATCH) return;
  float acc[NEXP];
#pragma unroll
  for (int e = 0; e < NEXP; ++e) acc[e] = gb[e];
  const float* xr = x + (size_t)b * FEAT;
  for (int i = 0; i < FEAT; ++i) {
    float xv = xr[i];
#pragma unroll
    for (int e = 0; e < NEXP; ++e) acc[e] += xv * gw[e * FEAT + i];
  }
  float m = acc[0];
#pragma unroll
  for (int e = 1; e < NEXP; ++e) m = fmaxf(m, acc[e]);
  float s = 0.f;
#pragma unroll
  for (int e = 0; e < NEXP; ++e) { acc[e] = __expf(acc[e] - m); s += acc[e]; }
  float inv = 1.0f / s;
#pragma unroll
  for (int e = 0; e < NEXP; ++e) gate[b * NEXP + e] = acc[e] * inv;
}

// ---------------- activation prep: [silu(v), B-spline bases(v)] -> bf16 ----------------
__global__ void prep_act_kernel(const float* __restrict__ act,
                                unsigned short* __restrict__ A) {
  int idx = blockIdx.x * blockDim.x + threadIdx.x;
  if (idx >= BATCH * FEAT) return;
  int b = idx / FEAT, f = idx % FEAT;
  float v = act[idx];
  float si = v / (1.0f + __expf(-v));           // silu

  // Cox-de Boor, order 3, extended grid g[t] = (t-3)*(2/5) - 1, t = 0..11
  const float h = 2.0f / GRIDN;
  float bas[GRIDN + 2 * SPLORD];                // 11 degree-0 bases
#pragma unroll
  for (int t = 0; t < GRIDN + 2 * SPLORD; ++t) {
    float g0 = (t     - SPLORD) * h - 1.0f;
    float g1 = (t + 1 - SPLORD) * h - 1.0f;
    bas[t] = (v >= g0 && v < g1) ? 1.0f : 0.0f;
  }
#pragma unroll
  for (int k = 1; k <= SPLORD; ++k) {
#pragma unroll
    for (int t = 0; t < GRIDN + 2 * SPLORD - k; ++t) {
      float gt   = (t         - SPLORD) * h - 1.0f;
      float gtk  = (t + k     - SPLORD) * h - 1.0f;
      float gt1  = (t + 1     - SPLORD) * h - 1.0f;
      float gtk1 = (t + k + 1 - SPLORD) * h - 1.0f;
      float left  = (v - gt)   / (gtk  - gt)  * bas[t];
      float right = (gtk1 - v) / (gtk1 - gt1) * bas[t + 1];
      bas[t] = left + right;
    }
  }
  unsigned short* dst = A + (size_t)b * KFUSE + (size_t)f * (NC + 1);
  dst[0] = f2bf(si);
#pragma unroll
  for (int c = 0; c < NC; ++c) dst[1 + c] = f2bf(bas[c]);
}

// ------- weight fusion (batched over experts in grid.z): (E,N,KFUSE) bf16 -------
__global__ void prep_w_kernel(const float* __restrict__ bw,   // (E, N, F)
                              const float* __restrict__ sw,   // (E, N, F, NC)
                              const float* __restrict__ sc,   // (E, N, F)
                              unsigned short* __restrict__ W) {
  int z = blockIdx.z;
  int idx = blockIdx.x * blockDim.x + threadIdx.x;
  if (idx >= FEAT * FEAT) return;
  int f = idx % FEAT;
  int n = idx / FEAT;
  size_t base = (size_t)z * FEAT * FEAT + idx;
  unsigned short* dst = W + ((size_t)z * FEAT + n) * KFUSE + (size_t)f * (NC + 1);
  dst[0] = f2bf(bw[base]);
  float s = sc[base];
  const float* sp = sw + base * NC;
#pragma unroll
  for (int c = 0; c < NC; ++c) dst[1 + c] = f2bf(sp[c] * s);
}

// ---------------- WMMA bf16 GEMM: C(M,N) = A(M,K) * W(N,K)^T ----------------
// Double-buffered LDS staged via GLOBAL_LOAD_ASYNC_TO_LDS_B128 (ASYNCcnt).
// Template: EPI==0 -> C[z] = A*W[z]^T (experts batched over gridDim.z)
//           EPI==1 -> C += gate[row,e] * (A*W^T)
//           MI = 16-row WMMA tiles per wave along M (2 -> 128x128, 4 -> 256x128)
#define BN 128
#define BK 32
#define LROW 40   // ushort elems per LDS row (80 B: 16B aligned, conflict-free stride)

template <int EPI, int MI>
__global__ void __launch_bounds__(256)
kan_gemm(const unsigned short* __restrict__ A,     // (M, K) bf16
         const unsigned short* __restrict__ Wall,  // (Z, N, K) bf16
         float* __restrict__ Call,                 // EPI0: (Z, M, N); EPI1: (M, N)
         const float* __restrict__ gate, int e,
         int M, int N, int K) {
  constexpr int BM = MI * 64;            // 4 waves along M, MI*16 rows each
  __shared__ unsigned short As[2][BM * LROW];
  __shared__ unsigned short Bs[2][BN * LROW];

  const int z = blockIdx.z;
  const unsigned short* W = Wall + (size_t)z * N * K;
  float* C = (EPI == 0) ? (Call + (size_t)z * M * N) : Call;

  const int tid  = threadIdx.x;
  const int lane = tid & 31;
  const int wave = tid >> 5;        // 0..7
  const int wm   = wave & 3;        // 4 waves along M
  const int wn   = wave >> 2;       // 2 waves along N (64 cols each)
  const int m0   = blockIdx.y * BM;
  const int n0   = blockIdx.x * BN;

  // tile staging: 256 threads; thread owns 32B segment(s) per matrix
  const int lr  = tid >> 1;         // base tile row 0..127
  const int lsg = tid & 1;          // segment 0/1
  const unsigned short* aSrc = A + (size_t)(m0 + lr) * K + lsg * 16;
  const unsigned short* wSrc = W + (size_t)(n0 + lr) * K + lsg * 16;
  const unsigned aOff = (unsigned)(uintptr_t)(&As[0][lr * LROW + lsg * 16]);
  const unsigned bOff = (unsigned)(uintptr_t)(&Bs[0][lr * LROW + lsg * 16]);
  const unsigned aStg = (unsigned)(BM * LROW * sizeof(unsigned short));
  const unsigned bStg = (unsigned)(BN * LROW * sizeof(unsigned short));
  const unsigned aRowBlk = (unsigned)(128 * LROW * sizeof(unsigned short));
  const size_t   gRowBlk = (size_t)128 * K;

  auto issue = [&](int s, int kt) {
    const unsigned short* ga = aSrc + (size_t)kt * BK;
    const unsigned short* gb = wSrc + (size_t)kt * BK;
    unsigned la = aOff + (unsigned)s * aStg;
    unsigned lb = bOff + (unsigned)s * bStg;
#pragma unroll
    for (int rb = 0; rb < BM / 128; ++rb) {        // A tile: BM rows
      async_b128(la + rb * aRowBlk,      ga + rb * gRowBlk);
      async_b128(la + rb * aRowBlk + 16, ga + rb * gRowBlk + 8);
    }
    async_b128(lb,      gb);                        // B tile: 128 rows
    async_b128(lb + 16, gb + 8);
  };

  v8f acc[MI][4];
#pragma unroll
  for (int mi = 0; mi < MI; ++mi)
#pragma unroll
    for (int ni = 0; ni < 4; ++ni)
#pragma unroll
      for (int r = 0; r < 8; ++r) acc[mi][ni][r] = 0.0f;

  const int aK = (lane >> 4) * 8;    // A frag K base (ISA 16-bit A layout)
  const int bK = (lane >> 4) * 8;    // B frag K base (ISA 16-bit layout, matches A)
  const int nr = lane & 15;

  const int T = K / BK;
  issue(0, 0);                       // prologue fill of stage 0
  for (int kt = 0; kt < T; ++kt) {
    const int cur = kt & 1;
    wait_async0();                   // my stage-cur writes have landed
    __syncthreads();                 // everyone's writes landed; prior reads retired
    if (kt + 1 < T) issue(1 - cur, kt + 1);   // fill next stage during compute

    const unsigned short* Ab = As[cur];
    const unsigned short* Bb = Bs[cur];
    v16bf aF[MI], bF[4];
#pragma unroll
    for (int mi = 0; mi < MI; ++mi) {
      const unsigned short* p = Ab + (wm * (MI * 16) + mi * 16 + nr) * LROW + aK;
      union { u32x4 q[2]; v16bf v; } u;
      u.q[0] = *reinterpret_cast<const u32x4*>(p);        // K = Kb..Kb+7
      u.q[1] = *reinterpret_cast<const u32x4*>(p + 16);   // K = Kb+16..Kb+23
      aF[mi] = u.v;
    }
#pragma unroll
    for (int ni = 0; ni < 4; ++ni) {
      const unsigned short* p = Bb + (wn * 64 + ni * 16 + nr) * LROW + bK;
      union { u32x4 q[2]; v16bf v; } u;
      u.q[0] = *reinterpret_cast<const u32x4*>(p);        // K = Kb..Kb+7
      u.q[1] = *reinterpret_cast<const u32x4*>(p + 16);   // K = Kb+16..Kb+23
      bF[ni] = u.v;
    }
#pragma unroll
    for (int mi = 0; mi < MI; ++mi)
#pragma unroll
      for (int ni = 0; ni < 4; ++ni)
        acc[mi][ni] = __builtin_amdgcn_wmma_f32_16x16x32_bf16(
            false, aF[mi], false, bF[ni], (short)0, acc[mi][ni], false, false);
  }

  // epilogue; D layout: VGPR r, lanes0-15 -> M=r, lanes16-31 -> M=8+r, N=lane&15
  const int mh = (lane >> 4) * 8;
#pragma unroll
  for (int mi = 0; mi < MI; ++mi) {
#pragma unroll
    for (int r = 0; r < 8; ++r) {
      int row = m0 + wm * (MI * 16) + mi * 16 + mh + r;
      if (EPI == 0) {
#pragma unroll
        for (int ni = 0; ni < 4; ++ni)
          C[(size_t)row * N + (n0 + wn * 64 + ni * 16 + nr)] = acc[mi][ni][r];
      } else {
        float g = gate[row * NEXP + e];
#pragma unroll
        for (int ni = 0; ni < 4; ++ni) {
          size_t off = (size_t)row * N + (n0 + wn * 64 + ni * 16 + nr);
          C[off] += g * acc[mi][ni][r];
        }
      }
    }
  }
}

// ---------------- host orchestration ----------------
extern "C" void kernel_launch(void* const* d_in, const int* in_sizes, int n_in,
                              void* d_out, int out_size, void* d_ws, size_t ws_size,
                              hipStream_t stream) {
  const float* x         = (const float*)d_in[0];
  const float* gate_w    = (const float*)d_in[1];
  const float* gate_b    = (const float*)d_in[2];
  const float* base_w1   = (const float*)d_in[3];
  const float* spline_w1 = (const float*)d_in[4];
  const float* scaler1   = (const float*)d_in[5];
  const float* base_w2   = (const float*)d_in[6];
  const float* spline_w2 = (const float*)d_in[7];
  const float* scaler2   = (const float*)d_in[8];
  float* out = (float*)d_out;
  (void)in_sizes; (void)n_in; (void)ws_size;

  // workspace carve-up (256B aligned)
  uintptr_t p = (uintptr_t)d_ws;
  auto al = [](uintptr_t v) { return (v + 255) & ~(uintptr_t)255; };
  float* gate_probs    = (float*)p;          p = al(p + (size_t)BATCH * NEXP * 4);
  unsigned short* A1   = (unsigned short*)p; p = al(p + (size_t)BATCH * KFUSE * 2);
  unsigned short* A2   = (unsigned short*)p; p = al(p + (size_t)BATCH * KFUSE * 2);
  unsigned short* W1a  = (unsigned short*)p; p = al(p + (size_t)NEXP * FEAT * KFUSE * 2);
  unsigned short* W2a  = (unsigned short*)p; p = al(p + (size_t)NEXP * FEAT * KFUSE * 2);
  float* hAll          = (float*)p;          p = al(p + (size_t)NEXP * BATCH * FEAT * 4);

  hipMemsetAsync(d_out, 0, (size_t)out_size * sizeof(float), stream);

  gate_kernel<<<(BATCH + 255) / 256, 256, 0, stream>>>(x, gate_w, gate_b, gate_probs);
  prep_act_kernel<<<(BATCH * FEAT + 255) / 256, 256, 0, stream>>>(x, A1);

  const dim3 wGrid((FEAT * FEAT + 255) / 256, 1, NEXP);
  prep_w_kernel<<<wGrid, 256, 0, stream>>>(base_w1, spline_w1, scaler1, W1a);
  prep_w_kernel<<<wGrid, 256, 0, stream>>>(base_w2, spline_w2, scaler2, W2a);

  // layer 1: all experts batched over grid.z; fat 256x128 tiles (16 WMMA / 16 ds_load)
  kan_gemm<0, 4><<<dim3(FEAT / BN, BATCH / 256, NEXP), 256, 0, stream>>>(
      A1, W1a, hAll, nullptr, 0, BATCH, FEAT, KFUSE);

  // layer 2: sequential per expert (reuses A2); 128x128 tiles for wider grid fill
  for (int e = 0; e < NEXP; ++e) {
    prep_act_kernel<<<(BATCH * FEAT + 255) / 256, 256, 0, stream>>>(
        hAll + (size_t)e * BATCH * FEAT, A2);
    kan_gemm<1, 2><<<dim3(FEAT / BN, BATCH / 128, 1), 256, 0, stream>>>(
        A2, W2a + (size_t)e * FEAT * KFUSE, out, gate_probs, e, BATCH, FEAT, KFUSE);
  }
}
